// Block_15479062135480
// MI455X (gfx1250) — matrix operand order repaired and checked by reading, not compiled, not run on hardware
//
#include <hip/hip_runtime.h>
#include <math.h>
#include <float.h>

typedef __attribute__((ext_vector_type(16))) _Float16 v16h;
typedef __attribute__((ext_vector_type(8)))  float    v8f;
typedef _Float16 h16;

#define B_   2
#define T_   2048
#define E_   1024
#define QKV_ 1024
#define H_   16
#define DH_  64
#define MT_  (B_*T_)          /* 4096 rows of activations */
#define WROWS_ (5*QKV_+E_)    /* 6144 weight rows        */
#define RMS_EPS 1.1920929e-7f

// ---------------------------------------------------------------- async copy
// global -> LDS direct DMA (CDNA5), 16 bytes per lane. Tracked by ASYNCcnt.
static __device__ inline void async_cp16(const void* g, void* l) {
  typedef __attribute__((address_space(3))) void lds_void;
  unsigned lo = (unsigned)(uintptr_t)(lds_void*)l;
  asm volatile("global_load_async_to_lds_b128 %0, %1, off"
               :: "v"(lo), "v"((unsigned long long)(uintptr_t)g)
               : "memory");
}
static __device__ inline void wait_async0() {
  asm volatile("s_wait_asynccnt 0x0" ::: "memory");
}

// ---------------------------------------------------------------- fragments
union FragU { v16h v; uint4 u[2]; };

static __device__ inline v8f vzero8() {
  v8f z;
  #pragma unroll
  for (int i = 0; i < 8; ++i) z[i] = 0.0f;
  return z;
}

// A fragment (16x32 f16, M-per-lane layout). m0: tile row, kb: k base in LDS row.
static __device__ inline v16h ld_frag_a(const h16* s, int stride, int m0, int kb) {
  const int lane = threadIdx.x & 31;
  const int m  = m0 + (lane & 15);
  const int kg = kb + ((lane >> 4) << 3);
  FragU f;
  f.u[0] = *(const uint4*)(s + m * stride + kg);
  f.u[1] = *(const uint4*)(s + m * stride + kg + 16);
  return f.v;
}
// B fragment (32x16 f16, N-per-lane layout). n0: tile col, kb: k base.
static __device__ inline v16h ld_frag_b(const h16* s, int stride, int n0, int kb) {
  const int lane = threadIdx.x & 31;
  const int n  = n0 + (lane & 15);
  const int kh = kb + ((lane >> 4) << 3);
  FragU f;
  f.u[0] = *(const uint4*)(s + n * stride + kh);
  f.u[1] = *(const uint4*)(s + n * stride + kh + 16);
  return f.v;
}
static __device__ inline v8f wmma16(v16h a, v16h b, v8f c) {
  return __builtin_amdgcn_wmma_f32_16x16x32_f16(false, a, false, b, (short)0, c, false, false);
}

// ---------------------------------------------------------------- reductions
static __device__ inline float blockReduceSum(float v, float* sbuf) {
  const int tid = threadIdx.x;
  sbuf[tid] = v; __syncthreads();
  for (int s = 128; s > 0; s >>= 1) {
    if (tid < s) sbuf[tid] += sbuf[tid + s];
    __syncthreads();
  }
  float r = sbuf[0]; __syncthreads();
  return r;
}
static __device__ inline float blockReduceMax(float v, float* sbuf) {
  const int tid = threadIdx.x;
  sbuf[tid] = v; __syncthreads();
  for (int s = 128; s > 0; s >>= 1) {
    if (tid < s) sbuf[tid] = fmaxf(sbuf[tid], sbuf[tid + s]);
    __syncthreads();
  }
  float r = sbuf[0]; __syncthreads();
  return r;
}

// ================================================================ GEMM
// C[M,N] = alpha * op(A) @ op(B)^T  (+ X residual).
// TRA==0: A is (M,K) row-major.  TRA==1: A is (K,M) row-major (computes A^T @ ..)
// TRB==0: Bt is (N,K) row-major. TRB==1: B is (K,N) row-major.
// M,N multiples of 128; K multiple of 64.
// Double-buffered LDS; non-transposed staging uses async global->LDS DMA.
template<int TRA, int TRB, int RESID>
__global__ __launch_bounds__(256)
void gemm_wmma(const h16* __restrict__ A, const h16* __restrict__ Bt,
               float* __restrict__ C, const float* __restrict__ X,
               const float* __restrict__ alpha_dev,
               int M, int N, int K, int lda, int ldb) {
  __shared__ h16 As[2][128][72];
  __shared__ h16 Bs[2][128][72];
  const int tid  = threadIdx.x;
  const int lane = tid & 31;
  const int wave = tid >> 5;
  const int wm = wave & 3;         // 4 row-waves
  const int wn = wave >> 2;        // 2 col-waves
  const int bm0 = blockIdx.y * 128;
  const int bn0 = blockIdx.x * 128;

  auto stage = [&](int k0, int buf) {
    #pragma unroll
    for (int i = 0; i < 4; ++i) {
      const int chunk = i * 256 + tid;               // 0..1023, 16B each
      if (!TRA) {
        const int r = chunk >> 3, c = (chunk & 7) * 8;
        async_cp16(A + (size_t)(bm0 + r) * lda + (k0 + c), &As[buf][r][c]);
      } else {
        const int kr = chunk >> 4, mc = (chunk & 15) * 8;
        uint4 d = *(const uint4*)(A + (size_t)(k0 + kr) * lda + (bm0 + mc));
        const h16* hp = (const h16*)&d;
        #pragma unroll
        for (int j = 0; j < 8; ++j) As[buf][mc + j][kr] = hp[j];
      }
      if (!TRB) {
        const int r = chunk >> 3, c = (chunk & 7) * 8;
        async_cp16(Bt + (size_t)(bn0 + r) * ldb + (k0 + c), &Bs[buf][r][c]);
      } else {
        const int kr = chunk >> 4, nc = (chunk & 15) * 8;
        uint4 d = *(const uint4*)(Bt + (size_t)(k0 + kr) * ldb + (bn0 + nc));
        const h16* hp = (const h16*)&d;
        #pragma unroll
        for (int j = 0; j < 8; ++j) Bs[buf][nc + j][kr] = hp[j];
      }
    }
  };

  v8f acc[2][4];
  #pragma unroll
  for (int i = 0; i < 2; ++i)
    #pragma unroll
    for (int j = 0; j < 4; ++j) acc[i][j] = vzero8();

  stage(0, 0);
  if (!TRA || !TRB) wait_async0();
  __syncthreads();

  int it = 0;
  for (int k0 = 0; k0 < K; k0 += 64, ++it) {
    const int cur = it & 1;
    const bool more = (k0 + 64) < K;
    if (more) stage(k0 + 64, cur ^ 1);   // overlap next tile with compute

    #pragma unroll
    for (int kb = 0; kb < 64; kb += 32) {
      v16h af[2], bf[4];
      #pragma unroll
      for (int i = 0; i < 2; ++i) af[i] = ld_frag_a(&As[cur][0][0], 72, wm * 32 + i * 16, kb);
      #pragma unroll
      for (int j = 0; j < 4; ++j) bf[j] = ld_frag_b(&Bs[cur][0][0], 72, wn * 64 + j * 16, kb);
      #pragma unroll
      for (int i = 0; i < 2; ++i)
        #pragma unroll
        for (int j = 0; j < 4; ++j) acc[i][j] = wmma16(af[i], bf[j], acc[i][j]);
    }
    if (more && (!TRA || !TRB)) wait_async0();
    __syncthreads();
  }

  const float alpha = alpha_dev ? alpha_dev[0] : 1.0f;
  const int mrow = (lane >> 4) << 3;
  #pragma unroll
  for (int i = 0; i < 2; ++i)
    #pragma unroll
    for (int j = 0; j < 4; ++j)
      #pragma unroll
      for (int r = 0; r < 8; ++r) {
        const int m = bm0 + wm * 32 + i * 16 + mrow + r;
        const int n = bn0 + wn * 64 + j * 16 + (lane & 15);
        float v = acc[i][j][r] * alpha;
        if (RESID) v += X[(size_t)m * N + n];
        C[(size_t)m * N + n] = v;
      }
}

// ================================================================ elementwise / norms
__global__ __launch_bounds__(256)
void convert_f32_f16(const float* __restrict__ in, h16* __restrict__ out, long n) {
  for (long i = (long)blockIdx.x * 256 + threadIdx.x; i < n; i += (long)gridDim.x * 256)
    out[i] = (h16)in[i];
}

__global__ __launch_bounds__(256)
void reduce_partial(const float* __restrict__ in, long n, float* __restrict__ partials) {
  __shared__ float s1[256], s2[256];
  float s = 0.f, ss = 0.f;
  for (long i = (long)blockIdx.x * 256 + threadIdx.x; i < n; i += (long)gridDim.x * 256) {
    float v = in[i]; s += v; ss += v * v;
  }
  s1[threadIdx.x] = s; s2[threadIdx.x] = ss; __syncthreads();
  for (int st = 128; st > 0; st >>= 1) {
    if (threadIdx.x < st) { s1[threadIdx.x] += s1[threadIdx.x + st]; s2[threadIdx.x] += s2[threadIdx.x + st]; }
    __syncthreads();
  }
  if (threadIdx.x == 0) { partials[blockIdx.x * 2] = s1[0]; partials[blockIdx.x * 2 + 1] = s2[0]; }
}

// scale = n_norm^-0.5 / (std(ddof=1) + 1e-8)
__global__ __launch_bounds__(256)
void finalize_scale(const float* __restrict__ partials, float n, float n_norm, float* __restrict__ out) {
  __shared__ float s1[256], s2[256];
  s1[threadIdx.x] = partials[threadIdx.x * 2];
  s2[threadIdx.x] = partials[threadIdx.x * 2 + 1];
  __syncthreads();
  for (int st = 128; st > 0; st >>= 1) {
    if (threadIdx.x < st) { s1[threadIdx.x] += s1[threadIdx.x + st]; s2[threadIdx.x] += s2[threadIdx.x + st]; }
    __syncthreads();
  }
  if (threadIdx.x == 0) {
    float sum = s1[0], sumsq = s2[0];
    float var = (sumsq - sum * sum / n) / (n - 1.0f);
    float sd = sqrtf(var);
    out[0] = (1.0f / sqrtf(n_norm)) / (sd + 1e-8f);
  }
}

// rmsnorm over rows of length `len`, input stride ld, f16 output stride ldo
__global__ __launch_bounds__(256)
void rmsnorm_rows(const float* __restrict__ in, int ld, h16* __restrict__ out, int ldo, int len) {
  __shared__ float sbuf[256];
  const float* p = in + (size_t)blockIdx.x * ld;
  float ss = 0.f;
  for (int i = threadIdx.x; i < len; i += 256) { float v = p[i]; ss += v * v; }
  ss = blockReduceSum(ss, sbuf);
  float r = rsqrtf(ss / (float)len + RMS_EPS);
  h16* o = out + (size_t)blockIdx.x * ldo;
  for (int i = threadIdx.x; i < len; i += 256) o[i] = (h16)(p[i] * r);
}

// aft: rmsnorm(q), rmsnorm(k); in-place -> sigmoid(qn), exp(kn), exp(kn)*v
__global__ __launch_bounds__(256)
void aft_qk(float* __restrict__ qkv) {
  __shared__ float sbuf[256];
  float* p = qkv + (size_t)blockIdx.x * (3 * QKV_);
  float sq = 0.f, sk = 0.f;
  for (int i = threadIdx.x; i < QKV_; i += 256) {
    float q = p[i];        sq += q * q;
    float k = p[QKV_ + i]; sk += k * k;
  }
  sq = blockReduceSum(sq, sbuf);
  sk = blockReduceSum(sk, sbuf);
  const float rq = rsqrtf(sq / (float)QKV_ + RMS_EPS);
  const float rk = rsqrtf(sk / (float)QKV_ + RMS_EPS);
  for (int i = threadIdx.x; i < QKV_; i += 256) {
    float q = p[i] * rq;
    p[i] = 1.0f / (1.0f + expf(-q));         // sigmoid(qn)
    float w = expf(p[QKV_ + i] * rk);        // exp(kn)
    p[QKV_ + i] = w;
    p[2 * QKV_ + i] = w * p[2 * QKV_ + i];   // exp(kn)*v
  }
}

// --- chunked cumsum over T (8 chunks of 256) ---
#define NCH 8
#define CHT 256
__global__ __launch_bounds__(256)
void csum_phase1(const float* __restrict__ qkv, float* __restrict__ cW, float* __restrict__ cWV) {
  const int e = blockIdx.x * 256 + threadIdx.x;
  const int ch = blockIdx.y, b = blockIdx.z;
  float sw = 0.f, swv = 0.f;
  for (int t = ch * CHT; t < (ch + 1) * CHT; ++t) {
    const size_t base = (size_t)(b * T_ + t) * (3 * QKV_);
    sw  += qkv[base + QKV_ + e];
    swv += qkv[base + 2 * QKV_ + e];
  }
  cW [(size_t)(b * NCH + ch) * QKV_ + e] = sw;
  cWV[(size_t)(b * NCH + ch) * QKV_ + e] = swv;
}
__global__ __launch_bounds__(256)
void csum_phase2(float* __restrict__ cW, float* __restrict__ cWV) {
  const int idx = blockIdx.x * 256 + threadIdx.x;   // B_*QKV_ threads
  const int b = idx >> 10, e = idx & (QKV_ - 1);
  float rw = 0.f, rwv = 0.f;
  for (int ch = 0; ch < NCH; ++ch) {
    const size_t o = (size_t)(b * NCH + ch) * QKV_ + e;
    float tw = cW[o], twv = cWV[o];
    cW[o] = rw; cWV[o] = rwv;
    rw += tw; rwv += twv;
  }
}
__global__ __launch_bounds__(256)
void csum_phase3(const float* __restrict__ qkv, const float* __restrict__ cW,
                 const float* __restrict__ cWV, h16* __restrict__ y16) {
  const int e = blockIdx.x * 256 + threadIdx.x;
  const int ch = blockIdx.y, b = blockIdx.z;
  float rw  = cW [(size_t)(b * NCH + ch) * QKV_ + e];
  float rwv = cWV[(size_t)(b * NCH + ch) * QKV_ + e];
  for (int t = ch * CHT; t < (ch + 1) * CHT; ++t) {
    const size_t base = (size_t)(b * T_ + t) * (3 * QKV_);
    rw  += qkv[base + QKV_ + e];
    rwv += qkv[base + 2 * QKV_ + e];
    float sig = qkv[base + e];
    y16[(size_t)(b * T_ + t) * QKV_ + e] = (h16)(sig * rwv / (rw + 1e-6f));
  }
}

// uv -> g = u*silu(v) (f16), a = v (f16)
__global__ __launch_bounds__(256)
void swiglu_post(const float* __restrict__ uv, h16* __restrict__ g16, h16* __restrict__ a16) {
  const size_t row = blockIdx.x;
  const float* p = uv + row * (2 * E_);
  for (int i = threadIdx.x; i < E_; i += 256) {
    float u = p[i], v = p[E_ + i];
    float s = v / (1.0f + expf(-v));
    g16[row * E_ + i] = (h16)(u * s);
    a16[row * E_ + i] = (h16)v;
  }
}

// softmax over rows of S (1024x1024) with scale E^-0.5; writes TRANSPOSED f16
__global__ __launch_bounds__(256)
void softmax_rows_T(const float* __restrict__ S, h16* __restrict__ outT) {
  __shared__ float sbuf[256];
  const int row = blockIdx.x;
  const float* p = S + (size_t)row * E_;
  const float sc = 0.03125f; // 1024^-0.5
  float mx = -INFINITY;
  for (int i = threadIdx.x; i < E_; i += 256) mx = fmaxf(mx, p[i] * sc);
  mx = blockReduceMax(mx, sbuf);
  float vals[4]; float sum = 0.f;
  #pragma unroll
  for (int c = 0; c < 4; ++c) {
    int i = threadIdx.x + c * 256;
    vals[c] = expf(p[i] * sc - mx); sum += vals[c];
  }
  sum = blockReduceSum(sum, sbuf);
  float inv = 1.0f / sum;
  #pragma unroll
  for (int c = 0; c < 4; ++c) {
    int i = threadIdx.x + c * 256;
    outT[(size_t)i * E_ + row] = (h16)(vals[c] * inv);
  }
}

// w = wT*scale*silu(w) + w
__global__ __launch_bounds__(256)
void weight_update(float* __restrict__ w, const float* __restrict__ wT,
                   const float* __restrict__ scale, long n) {
  const float sc = scale[0];
  for (long i = (long)blockIdx.x * 256 + threadIdx.x; i < n; i += (long)gridDim.x * 256) {
    float wv = w[i];
    float silu = wv / (1.0f + expf(-wv));
    w[i] = wT[i] * sc * silu + wv;
  }
}

// tea: rotary + rmsnorm(DH) on q,k; split to (B,H,T,DH) f16
__global__ __launch_bounds__(256)
void rotary_rms_split(const float* __restrict__ qkv, const float* __restrict__ cs,
                      const float* __restrict__ sn, h16* __restrict__ Qo,
                      h16* __restrict__ Ko, h16* __restrict__ Vo) {
  const int idx = blockIdx.x * 256 + threadIdx.x;   // B*T*H
  const int h = idx % H_;
  const int bt = idx / H_;
  const int t = bt % T_, b = bt / T_;
  const float* row = qkv + (size_t)bt * (3 * QKV_) + h * (3 * DH_);
  const float* cp = cs + t * 32; const float* sp = sn + t * 32;
  float q[DH_], k[DH_];
  #pragma unroll
  for (int j = 0; j < 32; ++j) {
    float c = cp[j], s = sp[j];
    float x1 = row[j], x2 = row[32 + j];
    q[j] = x1 * c + x2 * s; q[32 + j] = -x1 * s + x2 * c;
    float y1 = row[64 + j], y2 = row[96 + j];
    k[j] = y1 * c + y2 * s; k[32 + j] = -y1 * s + y2 * c;
  }
  float sq = 0.f, sk = 0.f;
  #pragma unroll
  for (int j = 0; j < DH_; ++j) { sq += q[j] * q[j]; sk += k[j] * k[j]; }
  const float rq = rsqrtf(sq / (float)DH_ + RMS_EPS);
  const float rk = rsqrtf(sk / (float)DH_ + RMS_EPS);
  const size_t ob = ((size_t)(b * H_ + h) * T_ + t) * DH_;
  #pragma unroll
  for (int j = 0; j < DH_; ++j) {
    Qo[ob + j] = (h16)(q[j] * rq);
    Ko[ob + j] = (h16)(k[j] * rk);
    Vo[ob + j] = (h16)row[128 + j];
  }
}

// ================================================================ flash attention
// grid (T/64, B*H), 128 threads (4 waves, 16 q-rows each). y16 out in (B,T,H*DH).
__global__ __launch_bounds__(128)
void flash_attn(const h16* __restrict__ Q, const h16* __restrict__ Kk,
                const h16* __restrict__ Vv, h16* __restrict__ Y) {
  __shared__ h16 Ks[64][72];
  __shared__ h16 Vs[64][72];          // transposed: [dh][key]
  __shared__ h16 Ps[4][16][72];       // per-wave probs
  const int tid = threadIdx.x, lane = tid & 31, wave = tid >> 5;
  const int bh = blockIdx.y, b = bh / H_, h = bh % H_;
  const int q0 = blockIdx.x * 64;
  const size_t base = (size_t)bh * T_ * DH_;
  const h16* qp = Q + base; const h16* kp = Kk + base; const h16* vp = Vv + base;
  const int mrow = (lane >> 4) << 3;

  FragU af[2];
  {
    const int m = q0 + wave * 16 + (lane & 15);
    const int kg = (lane >> 4) << 3;
    const h16* row = qp + (size_t)m * DH_;
    af[0].u[0] = *(const uint4*)(row + kg);
    af[0].u[1] = *(const uint4*)(row + kg + 16);
    af[1].u[0] = *(const uint4*)(row + 32 + kg);
    af[1].u[1] = *(const uint4*)(row + 32 + kg + 16);
  }
  float mrun[8], lrun[8];
  v8f acc[4];
  #pragma unroll
  for (int r = 0; r < 8; ++r) { mrun[r] = -INFINITY; lrun[r] = 0.f; }
  #pragma unroll
  for (int d = 0; d < 4; ++d) acc[d] = vzero8();

  for (int kt = 0; kt <= q0; kt += 64) {
    // stage K (async DMA, row-major) and V (transposed, manual)
    #pragma unroll
    for (int i = 0; i < 4; ++i) {
      const int chunk = i * 128 + tid;          // 0..511
      const int r = chunk >> 3, c = (chunk & 7) * 8;
      async_cp16(kp + (size_t)(kt + r) * DH_ + c, &Ks[r][c]);
      uint4 dv = *(const uint4*)(vp + (size_t)(kt + r) * DH_ + c);
      const h16* hp = (const h16*)&dv;
      #pragma unroll
      for (int j = 0; j < 8; ++j) Vs[c + j][r] = hp[j];
    }
    wait_async0();
    __syncthreads();

    // scores S = q @ k^T * DH^-0.5, causal mask
    float p[4][8];
    #pragma unroll
    for (int j = 0; j < 4; ++j) {
      v8f s = vzero8();
      s = wmma16(af[0].v, ld_frag_b(&Ks[0][0], 72, j * 16, 0), s);
      s = wmma16(af[1].v, ld_frag_b(&Ks[0][0], 72, j * 16, 32), s);
      const int kg = kt + j * 16 + (lane & 15);
      #pragma unroll
      for (int r = 0; r < 8; ++r) {
        const int qg = q0 + wave * 16 + mrow + r;
        p[j][r] = (kg > qg) ? -INFINITY : s[r] * 0.125f;
      }
    }
    // online softmax per row (16-lane halves hold one row)
    #pragma unroll
    for (int r = 0; r < 8; ++r) {
      float mx = fmaxf(fmaxf(p[0][r], p[1][r]), fmaxf(p[2][r], p[3][r]));
      mx = fmaxf(mx, __shfl_xor(mx, 1, 32));
      mx = fmaxf(mx, __shfl_xor(mx, 2, 32));
      mx = fmaxf(mx, __shfl_xor(mx, 4, 32));
      mx = fmaxf(mx, __shfl_xor(mx, 8, 32));
      const float mnew = fmaxf(mrun[r], mx);
      const float scl = expf(mrun[r] - mnew);
      float rs = 0.f;
      #pragma unroll
      for (int j = 0; j < 4; ++j) { p[j][r] = expf(p[j][r] - mnew); rs += p[j][r]; }
      rs += __shfl_xor(rs, 1, 32);
      rs += __shfl_xor(rs, 2, 32);
      rs += __shfl_xor(rs, 4, 32);
      rs += __shfl_xor(rs, 8, 32);
      lrun[r] = lrun[r] * scl + rs;
      mrun[r] = mnew;
      #pragma unroll
      for (int d = 0; d < 4; ++d) acc[d][r] = acc[d][r] * scl;
    }
    // C-layout -> A-layout via per-wave LDS
    #pragma unroll
    for (int j = 0; j < 4; ++j)
      #pragma unroll
      for (int r = 0; r < 8; ++r)
        Ps[wave][mrow + r][j * 16 + (lane & 15)] = (h16)p[j][r];
    // out += P @ V
    #pragma unroll
    for (int d = 0; d < 4; ++d) {
      acc[d] = wmma16(ld_frag_a(&Ps[wave][0][0], 72, 0, 0),  ld_frag_b(&Vs[0][0], 72, d * 16, 0),  acc[d]);
      acc[d] = wmma16(ld_frag_a(&Ps[wave][0][0], 72, 0, 32), ld_frag_b(&Vs[0][0], 72, d * 16, 32), acc[d]);
    }
    __syncthreads();
  }
  // write y (B,T,H*DH)
  #pragma unroll
  for (int d = 0; d < 4; ++d)
    #pragma unroll
    for (int r = 0; r < 8; ++r) {
      const int qg = q0 + wave * 16 + mrow + r;
      const int n = h * DH_ + d * 16 + (lane & 15);
      Y[(size_t)(b * T_ + qg) * QKV_ + n] = (h16)(acc[d][r] / lrun[r]);
    }
}

// ================================================================ host
extern "C" void kernel_launch(void* const* d_in, const int* in_sizes, int n_in,
                              void* d_out, int out_size, void* d_ws, size_t ws_size,
                              hipStream_t stream) {
  (void)in_sizes; (void)n_in; (void)out_size; (void)ws_size;
  const float* x_in = (const float*)d_in[0];
  const float* w_in = (const float*)d_in[1];
  const float* cosp = (const float*)d_in[2];
  const float* sinp = (const float*)d_in[3];

  char* ws = (char*)d_ws;
  size_t off = 0;
  auto alloc = [&](size_t bytes) -> void* {
    void* p = ws + off; off += (bytes + 255) & ~(size_t)255; return p;
  };
  float* w_cur  = (float*)alloc((size_t)WROWS_ * E_ * 4);         // 25.2 MB
  h16*   w16    = (h16*)  alloc((size_t)WROWS_ * E_ * 2);         // 12.6 MB
  float* big    = (float*)alloc((size_t)MT_ * 3 * QKV_ * 4);      // 50.3 MB (qkv/uv/wT)
  float* xa     = (float*)alloc((size_t)MT_ * E_ * 4);
  float* xb     = (float*)alloc((size_t)MT_ * E_ * 4);
  h16*   xn16   = (h16*)  alloc((size_t)MT_ * E_ * 2);
  h16*   y16    = (h16*)  alloc((size_t)MT_ * QKV_ * 2);
  h16*   g16    = (h16*)  alloc((size_t)MT_ * E_ * 2);            // also q16 (tea)
  h16*   o16    = (h16*)  alloc((size_t)MT_ * E_ * 2);            // also k16 (tea)
  h16*   a16    = (h16*)  alloc((size_t)MT_ * E_ * 2);            // also v16 (tea)
  float* oa_pre = (float*)alloc((size_t)E_ * E_ * 4);
  h16*   oa_t16 = (h16*)  alloc((size_t)E_ * E_ * 2);
  float* cW     = (float*)alloc((size_t)B_ * NCH * QKV_ * 4);
  float* cWV    = (float*)alloc((size_t)B_ * NCH * QKV_ * 4);
  float* parts  = (float*)alloc(256 * 2 * 4);
  float* scal   = (float*)alloc(16 * 4);

  hipMemcpyAsync(w_cur, w_in, (size_t)WROWS_ * E_ * 4, hipMemcpyDeviceToDevice, stream);

  const long WN = (long)WROWS_ * E_;
  const float* xcur = x_in;
  float* xout = xa;

  for (int layer = 0; layer < 2; ++layer) {
    convert_f32_f16<<<2048, 256, 0, stream>>>(w_cur, w16, WN);
    // w_norm scales
    reduce_partial<<<256, 256, 0, stream>>>(w_cur, (long)3 * QKV_ * E_, parts);
    finalize_scale<<<1, 256, 0, stream>>>(parts, 3.f * QKV_ * E_, (float)E_, scal + 0);
    reduce_partial<<<256, 256, 0, stream>>>(w_cur + (size_t)3 * QKV_ * E_, (long)2 * QKV_ * E_, parts);
    finalize_scale<<<1, 256, 0, stream>>>(parts, 2.f * QKV_ * E_, (float)QKV_, scal + 1);
    reduce_partial<<<256, 256, 0, stream>>>(w_cur + (size_t)5 * QKV_ * E_, (long)E_ * E_, parts);
    finalize_scale<<<1, 256, 0, stream>>>(parts, (float)E_ * E_, (float)E_, scal + 2);
    // qkv = rmsnorm(x) @ w_qkv^T
    rmsnorm_rows<<<MT_, 256, 0, stream>>>(xcur, E_, xn16, E_, E_);
    gemm_wmma<0, 0, 0><<<dim3(3 * QKV_ / 128, MT_ / 128), 256, 0, stream>>>(
        xn16, w16, big, nullptr, scal + 0, MT_, 3 * QKV_, E_, E_, E_);
    // aft
    aft_qk<<<MT_, 256, 0, stream>>>(big);
    csum_phase1<<<dim3(QKV_ / 256, NCH, B_), 256, 0, stream>>>(big, cW, cWV);
    csum_phase2<<<(B_ * QKV_) / 256, 256, 0, stream>>>(cW, cWV);
    csum_phase3<<<dim3(QKV_ / 256, NCH, B_), 256, 0, stream>>>(big, cW, cWV, y16);
    // swiglu
    gemm_wmma<0, 0, 0><<<dim3(2 * E_ / 128, MT_ / 128), 256, 0, stream>>>(
        y16, w16 + (size_t)3 * QKV_ * E_, big, nullptr, scal + 1, MT_, 2 * E_, QKV_, QKV_, QKV_);
    rmsnorm_rows<<<MT_, 256, 0, stream>>>(big, 2 * E_, o16, E_, E_);
    swiglu_post<<<MT_, 256, 0, stream>>>(big, g16, a16);
    gemm_wmma<0, 0, 1><<<dim3(E_ / 128, MT_ / 128), 256, 0, stream>>>(
        g16, w16 + (size_t)5 * QKV_ * E_, xout, xcur, scal + 2, MT_, E_, E_, E_, E_);
    // oa = softmax(o^T @ a * E^-0.5)
    gemm_wmma<1, 1, 0><<<dim3(E_ / 128, E_ / 128), 256, 0, stream>>>(
        o16, a16, oa_pre, nullptr, nullptr, E_, E_, MT_, E_, E_);
    softmax_rows_T<<<E_, 256, 0, stream>>>(oa_pre, oa_t16);
    // wT = w_norm(w @ oa); w update
    gemm_wmma<0, 0, 0><<<dim3(E_ / 128, WROWS_ / 128), 256, 0, stream>>>(
        w16, oa_t16, big, nullptr, nullptr, WROWS_, E_, E_, E_, E_);
    reduce_partial<<<256, 256, 0, stream>>>(big, WN, parts);
    finalize_scale<<<1, 256, 0, stream>>>(parts, (float)WN, (float)WROWS_, scal + 3);
    weight_update<<<2048, 256, 0, stream>>>(w_cur, big, scal + 3, WN);

    xcur = xout;
    xout = (layer == 0) ? xb : xa;
  }

  // ---------------- final: tea + swiglu (no w_norm, no oa) ----------------
  convert_f32_f16<<<2048, 256, 0, stream>>>(w_cur, w16, WN);
  rmsnorm_rows<<<MT_, 256, 0, stream>>>(xcur, E_, xn16, E_, E_);
  gemm_wmma<0, 0, 0><<<dim3(3 * QKV_ / 128, MT_ / 128), 256, 0, stream>>>(
      xn16, w16, big, nullptr, nullptr, MT_, 3 * QKV_, E_, E_, E_);
  rotary_rms_split<<<(B_ * T_ * H_) / 256, 256, 0, stream>>>(big, cosp, sinp, g16, o16, a16);
  flash_attn<<<dim3(T_ / 64, B_ * H_), 128, 0, stream>>>(g16, o16, a16, y16);
  gemm_wmma<0, 0, 0><<<dim3(2 * E_ / 128, MT_ / 128), 256, 0, stream>>>(
      y16, w16 + (size_t)3 * QKV_ * E_, big, nullptr, nullptr, MT_, 2 * E_, QKV_, QKV_, QKV_);
  swiglu_post<<<MT_, 256, 0, stream>>>(big, g16, a16);
  gemm_wmma<0, 0, 1><<<dim3(E_ / 128, MT_ / 128), 256, 0, stream>>>(
      g16, w16 + (size_t)5 * QKV_ * E_, (float*)d_out, xcur, nullptr, MT_, E_, E_, E_, E_);
}
